// MultiHead_33414845563380
// MI455X (gfx1250) — matrix operand order repaired and checked by reading, not compiled, not run on hardware
//
#include <hip/hip_runtime.h>
#include <hip/hip_bf16.h>

// Anomaly-Transformer style multi-head attention for MI455X (gfx1250).
// GEMMs use v_wmma_f32_16x16x32_f16 (f16 A/B, f32 accumulate). The
// scores -> softmax -> series@V chain is fused per 16-row band through LDS so
// the 134MB series tensor is written exactly once and never re-read from HBM.

typedef _Float16 half8  __attribute__((ext_vector_type(8)));
typedef _Float16 half16 __attribute__((ext_vector_type(16)));
typedef float    f32x4  __attribute__((ext_vector_type(4)));
typedef float    f32x8  __attribute__((ext_vector_type(8)));

#define B_   4
#define L_   1024
#define H_   8
#define D_   512
#define DK_  64
#define N_   (B_ * L_)          // 4096 rows
#define HB_  (H_ * B_)          // 32
#define LL_  ((size_t)L_ * L_)  // 1048576

#define SROW 1028               // padded LDS row stride (floats): bank-conflict-free
#define OROW 68                 // padded LDS row stride for O partials

// ---------------- workspace layout (bytes) ----------------
#define OFF_Q16   0u                         // 4096*512 f16  = 4 MB
#define OFF_K16   4194304u
#define OFF_V16   8388608u
#define OFF_WQ16  12582912u                  // 512*512 f16 = 512 KB each
#define OFF_WK16  13107200u
#define OFF_WV16  13631488u
#define OFF_WO16  14155776u
#define OFF_QH16  14680064u                  // [HB][L][64] f16 = 4 MB
#define OFF_KH16  18874368u
#define OFF_VHT16 23068672u                  // [HB][64][L] f16 = 4 MB (transposed V)
#define OFF_O16   27262976u                  // [B*L][512] f16 = 4 MB (merged heads)
#define OFF_SH    31457280u                  // [HB][L] f32 = 128 KB
// total 31588352 bytes (~30.1 MB)

// ---- CDNA5 WMMA operand loaders (ISA 7.12.2 wave32 layouts) ----
// A 16x32 f16: lane l<16 -> M=l, K={kk..kk+7, kk+16..kk+23};
//              lane l>=16 -> M=l-16, K offsets +8.
static __device__ __forceinline__ half16 ldA16(const _Float16* __restrict__ row,
                                               int kk, int lane) {
  const int hs = (lane >= 16) ? 8 : 0;
  half8 lo = *(const half8*)(row + kk + hs);
  half8 hi = *(const half8*)(row + kk + 16 + hs);
  half16 a;
#pragma unroll
  for (int e = 0; e < 8; ++e) { a[e] = lo[e]; a[8 + e] = hi[e]; }
  return a;
}

// B 32x16 f16: lane holds column N=lane%16; K = {kk..kk+7, kk+16..kk+23} (+8 for lane>=16).
// 'colrun' points at the contiguous K-run for this lane's column.
static __device__ __forceinline__ half16 ldB16(const _Float16* __restrict__ colrun,
                                               int kk, int lane) {
  const int kb = kk + ((lane >= 16) ? 8 : 0);
  half8 lo = *(const half8*)(colrun + kb);
  half8 hi = *(const half8*)(colrun + kb + 16);
  half16 b;
#pragma unroll
  for (int e = 0; e < 8; ++e) { b[e] = lo[e]; b[8 + e] = hi[e]; }
  return b;
}

// Build f16 A-fragment from an fp32 row (LDS or global generic pointer).
static __device__ __forceinline__ half16 ldA32cvt(const float* row, int kk, int lane) {
  const int hs = (lane >= 16) ? 8 : 0;
  f32x4 s0 = *(const f32x4*)(row + kk + hs);
  f32x4 s1 = *(const f32x4*)(row + kk + hs + 4);
  f32x4 s2 = *(const f32x4*)(row + kk + 16 + hs);
  f32x4 s3 = *(const f32x4*)(row + kk + 16 + hs + 4);
  half16 a;
#pragma unroll
  for (int e = 0; e < 4; ++e) {
    a[e]      = (_Float16)s0[e];
    a[4 + e]  = (_Float16)s1[e];
    a[8 + e]  = (_Float16)s2[e];
    a[12 + e] = (_Float16)s3[e];
  }
  return a;
}

// ---------------- elementwise fp32 -> f16 ----------------
__global__ void k_cvt_f16(const float* __restrict__ src, _Float16* __restrict__ dst, int n) {
  int i = blockIdx.x * 256 + threadIdx.x;
  if (i < n) dst[i] = (_Float16)src[i];
}

// ---------------- sigma projection: Sh[h*B+b][l] = (Sigma @ Wsig^T)[b,l,h] ----------------
__global__ void k_sigma(const float* __restrict__ Sigma, const float* __restrict__ Wsig,
                        float* __restrict__ Sh) {
  int idx = blockIdx.x * 256 + threadIdx.x;       // 0 .. N_*H_-1
  if (idx >= N_ * H_) return;
  int h = idx & (H_ - 1);
  int n = idx >> 3;                                // b*L + l
  float acc = 0.f;
#pragma unroll
  for (int k = 0; k < H_; ++k) acc += Sigma[n * H_ + k] * Wsig[h * H_ + k];
  int b = n >> 10, l = n & (L_ - 1);
  Sh[((size_t)(h * B_ + b)) * L_ + l] = acc;
}

// ---------------- fused QKV projection: Y = X16 @ W16^T, head-split stores ----------------
// which=0: Qh [hb][l][64]; which=1: Kh [hb][l][64]; which=2: VhT [hb][d][L] (transposed).
__global__ void k_proj_qkv(const _Float16* __restrict__ Q16, const _Float16* __restrict__ K16,
                           const _Float16* __restrict__ V16, const _Float16* __restrict__ Wq,
                           const _Float16* __restrict__ Wk, const _Float16* __restrict__ Wv,
                           _Float16* __restrict__ Qh, _Float16* __restrict__ Kh,
                           _Float16* __restrict__ VhT) {
  const int lane = threadIdx.x;
  const int mt = blockIdx.x * 16;        // row tile (b*L+l)
  const int nt = blockIdx.y * 16;        // output column tile
  const int which = blockIdx.z;
  const _Float16* X = (which == 0) ? Q16 : (which == 1) ? K16 : V16;
  const _Float16* W = (which == 0) ? Wq  : (which == 1) ? Wk  : Wv;
  const _Float16* arow = X + (size_t)(mt + (lane & 15)) * D_;
  const _Float16* brun = W + (size_t)(nt + (lane & 15)) * D_;  // row j of W = K-run of B[:,j]
  f32x8 acc = {};
#pragma unroll 4
  for (int kk = 0; kk < D_; kk += 32) {
    half16 a = ldA16(arow, kk, lane);
    half16 b = ldB16(brun, kk, lane);
    acc = __builtin_amdgcn_wmma_f32_16x16x32_f16(false, a, false, b, (short)0, acc,
                                                 false, false);
  }
  const int col = nt + (lane & 15);
  const int h = col >> 6, d = col & 63;
  const int mbase = mt + ((lane >= 16) ? 8 : 0);
#pragma unroll
  for (int r = 0; r < 8; ++r) {
    const int m = mbase + r;             // b*L + l
    const int b = m >> 10, l = m & (L_ - 1);
    const size_t hb = (size_t)(h * B_ + b);
    const _Float16 v = (_Float16)acc[r];
    if (which == 2)      VhT[(hb * DK_ + d) * L_ + l] = v;
    else if (which == 1) Kh[(hb * L_ + l) * DK_ + d] = v;
    else                 Qh[(hb * L_ + l) * DK_ + d] = v;
  }
}

// ---------------- fused attention band: scores -> softmax -> series@V ----------------
// One 512-thread workgroup (16 waves) per (hb, 16-row band).
//   Phase 1: wave w computes score tiles for columns [w*64, w*64+64) into LDS.
//   Phase 2: wave w softmaxes LDS row w, writes the final series row to HBM once.
//   Phase 3: wave w (dt=w&3, kc=w>>2) computes a K-partial of the 16x64 O tile
//            from LDS series; partials combined in fixed order (deterministic).
__global__ void k_attn_band(const _Float16* __restrict__ Qh, const _Float16* __restrict__ Kh,
                            const _Float16* __restrict__ VhT, float* __restrict__ series,
                            _Float16* __restrict__ O16) {
  __shared__ __align__(16) float sS[16 * SROW];      // 16 x 1024 scores (padded)
  __shared__ __align__(16) float sO[4 * 16 * OROW];  // 4 K-partials of 16x64 O tile

  const int tid = threadIdx.x;
  const int lane = tid & 31;
  const int w = tid >> 5;                 // wave id 0..15
  const int i0 = blockIdx.x * 16;         // band row base
  const int hb = blockIdx.y;
  const int c16 = lane & 15;
  const int hi8 = (lane >= 16) ? 8 : 0;

  // ---- Phase 1: scores = (Qh Kh^T) * 0.125 for this band ----
  {
    const _Float16* arow = Qh + ((size_t)hb * L_ + i0 + c16) * DK_;
    const _Float16* b0 = Kh + ((size_t)hb * L_ + (w * 4 + 0) * 16 + c16) * DK_;
    const _Float16* b1 = Kh + ((size_t)hb * L_ + (w * 4 + 1) * 16 + c16) * DK_;
    const _Float16* b2 = Kh + ((size_t)hb * L_ + (w * 4 + 2) * 16 + c16) * DK_;
    const _Float16* b3 = Kh + ((size_t)hb * L_ + (w * 4 + 3) * 16 + c16) * DK_;
    f32x8 a0 = {}, a1 = {}, a2 = {}, a3 = {};
#pragma unroll
    for (int kk = 0; kk < DK_; kk += 32) {
      half16 af = ldA16(arow, kk, lane);
      a0 = __builtin_amdgcn_wmma_f32_16x16x32_f16(false, af, false, ldB16(b0, kk, lane),
                                                  (short)0, a0, false, false);
      a1 = __builtin_amdgcn_wmma_f32_16x16x32_f16(false, af, false, ldB16(b1, kk, lane),
                                                  (short)0, a1, false, false);
      a2 = __builtin_amdgcn_wmma_f32_16x16x32_f16(false, af, false, ldB16(b2, kk, lane),
                                                  (short)0, a2, false, false);
      a3 = __builtin_amdgcn_wmma_f32_16x16x32_f16(false, af, false, ldB16(b3, kk, lane),
                                                  (short)0, a3, false, false);
    }
#pragma unroll
    for (int r = 0; r < 8; ++r) {
      const int row = r + hi8;
      sS[row * SROW + (w * 4 + 0) * 16 + c16] = a0[r] * 0.125f;
      sS[row * SROW + (w * 4 + 1) * 16 + c16] = a1[r] * 0.125f;
      sS[row * SROW + (w * 4 + 2) * 16 + c16] = a2[r] * 0.125f;
      sS[row * SROW + (w * 4 + 3) * 16 + c16] = a3[r] * 0.125f;
    }
  }
  __syncthreads();

  // ---- Phase 2: softmax row w; write series row to HBM (single write) ----
  {
    float* prow = sS + w * SROW;
    float m = -3.402823466e38f;
#pragma unroll
    for (int t = 0; t < 32; ++t) m = fmaxf(m, prow[t * 32 + lane]);
#pragma unroll
    for (int o = 16; o > 0; o >>= 1) m = fmaxf(m, __shfl_xor(m, o, 32));
    float s = 0.f;
#pragma unroll
    for (int t = 0; t < 32; ++t) {
      const float e = expf(prow[t * 32 + lane] - m);
      prow[t * 32 + lane] = e;
      s += e;
    }
#pragma unroll
    for (int o = 16; o > 0; o >>= 1) s += __shfl_xor(s, o, 32);
    const float inv = 1.0f / s;
    float* grow = series + (size_t)hb * LL_ + (size_t)(i0 + w) * L_;
#pragma unroll
    for (int t = 0; t < 32; ++t) {
      const float v = prow[t * 32 + lane] * inv;
      prow[t * 32 + lane] = v;
      grow[t * 32 + lane] = v;
    }
  }
  __syncthreads();

  // ---- Phase 3: O(16x64) = series_band(16x1024) @ Vh(1024x64), K split 4 ways ----
  {
    const int dt = (w & 3) * 16;           // d-tile columns
    const int kc = (w >> 2) * 256;         // K-chunk base
    const _Float16* brun = VhT + ((size_t)hb * DK_ + dt + c16) * L_;
    const float* aLds = sS + c16 * SROW;
    f32x8 oacc = {};
#pragma unroll 4
    for (int kk = kc; kk < kc + 256; kk += 32) {
      half16 af = ldA32cvt(aLds, kk, lane);
      half16 bf = ldB16(brun, kk, lane);
      oacc = __builtin_amdgcn_wmma_f32_16x16x32_f16(false, af, false, bf, (short)0, oacc,
                                                    false, false);
    }
#pragma unroll
    for (int r = 0; r < 8; ++r)
      sO[((w >> 2) * 16 + r + hi8) * OROW + dt + c16] = oacc[r];
  }
  __syncthreads();

  // ---- combine K-partials in fixed order, store merged-head f16 O ----
  {
    const int h = hb >> 2, bb = hb & 3;
#pragma unroll
    for (int e = 0; e < 2; ++e) {
      const int elem = tid + e * 512;      // 0..1023
      const int row = elem >> 6, col = elem & 63;
      const float v = sO[(0 * 16 + row) * OROW + col] + sO[(1 * 16 + row) * OROW + col] +
                      sO[(2 * 16 + row) * OROW + col] + sO[(3 * 16 + row) * OROW + col];
      O16[((size_t)(bb * L_ + i0 + row)) * D_ + (h << 6) + col] = (_Float16)v;
    }
  }
}

// ---------------- Gaussian distance prior rows ----------------
__global__ void k_prior(const float* __restrict__ Sh, float* __restrict__ P) {
  __shared__ float red[256];
  const int t = threadIdx.x;
  const int row = blockIdx.x;              // hb*L + i
  const int hb = row >> 10, i = row & (L_ - 1);
  const float sg = Sh[(size_t)hb * L_ + i];
  const float inv = 0.3989422804014327f / sg;   // 1/(sqrt(2*pi)*sigma)
  const float c = 0.5f / (sg * sg);
  float z[4]; float ls = 0.f;
#pragma unroll
  for (int e = 0; e < 4; ++e) {
    const float d = (float)(i - (t * 4 + e));
    z[e] = inv * expf(-(d * d) * c);
    ls += z[e];
  }
  red[t] = ls; __syncthreads();
#pragma unroll
  for (int s = 128; s > 0; s >>= 1) {
    if (t < s) red[t] += red[t + s];
    __syncthreads();
  }
  const float rinv = 1.0f / red[0];
  f32x4 o; o[0] = z[0] * rinv; o[1] = z[1] * rinv; o[2] = z[2] * rinv; o[3] = z[3] * rinv;
  *(f32x4*)(P + (size_t)row * L_ + t * 4) = o;
}

// ---------------- out = O16 @ Wo^T (fp32 result) ----------------
__global__ void k_outproj(const _Float16* __restrict__ O16, const _Float16* __restrict__ Wo,
                          float* __restrict__ out) {
  const int lane = threadIdx.x;
  const int mt = blockIdx.x * 16, nt = blockIdx.y * 16;
  const _Float16* arow = O16 + (size_t)(mt + (lane & 15)) * D_;
  const _Float16* brun = Wo  + (size_t)(nt + (lane & 15)) * D_;
  f32x8 acc = {};
#pragma unroll 4
  for (int kk = 0; kk < D_; kk += 32) {
    half16 a = ldA16(arow, kk, lane);
    half16 b = ldB16(brun, kk, lane);
    acc = __builtin_amdgcn_wmma_f32_16x16x32_f16(false, a, false, b, (short)0, acc,
                                                 false, false);
  }
  const int n = nt + (lane & 15);
  const int mbase = mt + ((lane >= 16) ? 8 : 0);
#pragma unroll
  for (int r = 0; r < 8; ++r)
    out[(size_t)(mbase + r) * D_ + n] = acc[r];
}

extern "C" void kernel_launch(void* const* d_in, const int* in_sizes, int n_in,
                              void* d_out, int out_size, void* d_ws, size_t ws_size,
                              hipStream_t stream) {
  (void)in_sizes; (void)n_in; (void)out_size; (void)ws_size;
  const float* Sigma = (const float*)d_in[0];
  const float* Q     = (const float*)d_in[1];
  const float* K     = (const float*)d_in[2];
  const float* V     = (const float*)d_in[3];
  const float* Wsig  = (const float*)d_in[4];
  const float* Wq    = (const float*)d_in[5];
  const float* Wk    = (const float*)d_in[6];
  const float* Wv    = (const float*)d_in[7];
  const float* Wo    = (const float*)d_in[8];

  char* ws = (char*)d_ws;
  _Float16* Q16   = (_Float16*)(ws + OFF_Q16);
  _Float16* K16   = (_Float16*)(ws + OFF_K16);
  _Float16* V16   = (_Float16*)(ws + OFF_V16);
  _Float16* Wq16  = (_Float16*)(ws + OFF_WQ16);
  _Float16* Wk16  = (_Float16*)(ws + OFF_WK16);
  _Float16* Wv16  = (_Float16*)(ws + OFF_WV16);
  _Float16* Wo16  = (_Float16*)(ws + OFF_WO16);
  _Float16* Qh16  = (_Float16*)(ws + OFF_QH16);
  _Float16* Kh16  = (_Float16*)(ws + OFF_KH16);
  _Float16* VhT16 = (_Float16*)(ws + OFF_VHT16);
  _Float16* O16   = (_Float16*)(ws + OFF_O16);
  float*    Sh    = (float*)(ws + OFF_SH);

  float* prior  = (float*)d_out;                         // [32,1024,1024]
  float* series = (float*)d_out + (size_t)HB_ * LL_;     // [32,1024,1024]
  float* outp   = (float*)d_out + 2 * (size_t)HB_ * LL_; // [4,1024,512]

  const int nX = N_ * D_;     // 2097152
  const int nW = D_ * D_;     // 262144
  k_cvt_f16<<<(nX + 255) / 256, 256, 0, stream>>>(Q,  Q16,  nX);
  k_cvt_f16<<<(nX + 255) / 256, 256, 0, stream>>>(K,  K16,  nX);
  k_cvt_f16<<<(nX + 255) / 256, 256, 0, stream>>>(V,  V16,  nX);
  k_cvt_f16<<<(nW + 255) / 256, 256, 0, stream>>>(Wq, Wq16, nW);
  k_cvt_f16<<<(nW + 255) / 256, 256, 0, stream>>>(Wk, Wk16, nW);
  k_cvt_f16<<<(nW + 255) / 256, 256, 0, stream>>>(Wv, Wv16, nW);
  k_cvt_f16<<<(nW + 255) / 256, 256, 0, stream>>>(Wo, Wo16, nW);

  k_sigma<<<(N_ * H_ + 255) / 256, 256, 0, stream>>>(Sigma, Wsig, Sh);

  dim3 gProj(N_ / 16, D_ / 16, 3);
  k_proj_qkv<<<gProj, 32, 0, stream>>>(Q16, K16, V16, Wq16, Wk16, Wv16,
                                       Qh16, Kh16, VhT16);

  dim3 gBand(L_ / 16, HB_);
  k_attn_band<<<gBand, 512, 0, stream>>>(Qh16, Kh16, VhT16, series, O16);

  k_prior<<<HB_ * L_, 256, 0, stream>>>(Sh, prior);

  dim3 gOut(N_ / 16, D_ / 16);
  k_outproj<<<gOut, 32, 0, stream>>>(O16, Wo16, outp);
}
